// VGAEModel_32856499814660
// MI455X (gfx1250) — matrix operand order repaired and checked by reading, not compiled, not run on hardware
//
#include <hip/hip_runtime.h>
#include <hip/hip_bf16.h>
#include <cstdint>

#define IN_C  256
#define HID_C 256
#define OUT_C 128

typedef __attribute__((ext_vector_type(16))) __bf16 v16bf;
typedef __attribute__((ext_vector_type(8)))  __bf16 v8bf;
typedef __attribute__((ext_vector_type(4)))  __bf16 v4bf;
typedef __attribute__((ext_vector_type(8)))  float  v8f;

__device__ __forceinline__ void atomic_add_f32(float* p, float v) {
    __hip_atomic_fetch_add(p, v, __ATOMIC_RELAXED, __HIP_MEMORY_SCOPE_AGENT);
}

// ---------------- elementwise helpers ----------------

__global__ void k_zero_f32(float* p, long n) {
    long i = (long)blockIdx.x * blockDim.x + threadIdx.x;
    if (i < n) p[i] = 0.0f;
}

// vectorized f32 -> bf16 (n must be a multiple of 4; true for all N*256 / N*128 / weights)
__global__ void k_f32_to_bf16_v4(const float4* __restrict__ src, v4bf* __restrict__ dst, long n4) {
    long i = (long)blockIdx.x * blockDim.x + threadIdx.x;
    if (i >= n4) return;
    float4 f = src[i];
    v4bf o;
    o[0] = (__bf16)f.x; o[1] = (__bf16)f.y; o[2] = (__bf16)f.z; o[3] = (__bf16)f.w;
    dst[i] = o;
}

// deg[col[e]] += 1  (in-degree over target nodes)
__global__ void k_deg_count(const int* __restrict__ col, float* __restrict__ deg, int E) {
    int e = blockIdx.x * blockDim.x + threadIdx.x;
    if (e < E) atomic_add_f32(&deg[col[e]], 1.0f);
}

// dinv[v] = rsqrt(deg[v] + 1)   (+1 = self loop)
__global__ void k_dinv(const float* __restrict__ deg, float* __restrict__ dinv, int n) {
    int v = blockIdx.x * blockDim.x + threadIdx.x;
    if (v < n) dinv[v] = rsqrtf(deg[v] + 1.0f);
}

// for each edge (r -> c): out[c][:] += dinv[r]*dinv[c] * lin[r][:]   (4 channels / thread)
__global__ void k_agg_edges(const int* __restrict__ rows, const int* __restrict__ cols,
                            const float* __restrict__ dinv, const float* __restrict__ lin,
                            float* __restrict__ out, long total, int c4shift, int c4mask, int C) {
    long i = (long)blockIdx.x * blockDim.x + threadIdx.x;
    if (i >= total) return;
    int e  = (int)(i >> c4shift);
    int ch = (int)(i & c4mask);
    int r = rows[e], c = cols[e];
    float nrm = dinv[r] * dinv[c];
    const float4 h4 = *(const float4*)(lin + (size_t)r * C + ch * 4);
    float* o = out + (size_t)c * C + ch * 4;
    atomic_add_f32(o + 0, nrm * h4.x);
    atomic_add_f32(o + 1, nrm * h4.y);
    atomic_add_f32(o + 2, nrm * h4.z);
    atomic_add_f32(o + 3, nrm * h4.w);
}

// ---------------- WMMA GEMM with fused self-loop/bias epilogue --------------------------
// Cl[M,NC]    = A[M,K] @ B[K,NC]                       (kept for edge gathers)
// Cinit[M,NC] = dinv[m]^2 * Cl[m,n] + bias[n]          (initialized aggregation output)
// 8 waves/block; wave = one 16-row M tile x 64-col N group (4 accumulators).
// Double-buffered K loop: prefetch next A+B fragments while current WMMAs issue.
// waves_per_eu(4,8): allow ~256 VGPRs so the 2-deep fragment buffer stays in registers.
template <int K, int NC>
__global__ __launch_bounds__(256)
__attribute__((amdgpu_waves_per_eu(4, 8)))
void k_gemm_bf16_wmma(const __bf16* __restrict__ A, const __bf16* __restrict__ B,
                      float* __restrict__ Cl, float* __restrict__ Cinit,
                      const float* __restrict__ dinv, const float* __restrict__ bias,
                      int M) {
    constexpr int KSTEPS = K / 32;
    int wave = threadIdx.x >> 5;
    int lane = threadIdx.x & 31;
    int mTile = blockIdx.x * 8 + wave;
    if (mTile * 16 >= M) return;               // whole-wave early-out, EXEC stays all-ones

    int l16 = lane & 15;
    int sel = lane >> 4;                       // 0: lanes 0-15, 1: lanes 16-31
    int nBase = blockIdx.y * 64;

    const __bf16* ap = A + (size_t)(mTile * 16 + l16) * K + sel * 8;
    const __bf16* bp = B + nBase + l16;

    v8bf  alo[2], ahi[2];
    v16bf bfrag[2][4];
    v8f   acc[4] = {};

    // prefetch K-step 0
    alo[0] = *(const v8bf*)(ap);
    ahi[0] = *(const v8bf*)(ap + 16);
#pragma unroll
    for (int nt = 0; nt < 4; ++nt) for (int i = 0; i < 8; ++i) { bfrag[0][nt][i] = bp[(size_t)(sel * 8 + i) * NC + nt * 16]; bfrag[0][nt][i + 8] = bp[(size_t)(16 + sel * 8 + i) * NC + nt * 16]; }

#pragma unroll
    for (int ks = 0; ks < KSTEPS; ++ks) {
        const int cur = ks & 1;
        const int nxt = cur ^ 1;
        if (ks + 1 < KSTEPS) {                 // prefetch next K-step (immediate offsets)
            const int kb = (ks + 1) * 32;
            alo[nxt] = *(const v8bf*)(ap + kb);
            ahi[nxt] = *(const v8bf*)(ap + kb + 16);
            const __bf16* bb = bp + (size_t)kb * NC;
#pragma unroll
            for (int nt = 0; nt < 4; ++nt) for (int i = 0; i < 8; ++i) { bfrag[nxt][nt][i] = bb[(size_t)(sel * 8 + i) * NC + nt * 16]; bfrag[nxt][nt][i + 8] = bb[(size_t)(16 + sel * 8 + i) * NC + nt * 16]; }
        }
        v16bf a;
#pragma unroll
        for (int i = 0; i < 8; ++i) { a[i] = alo[cur][i]; a[i + 8] = ahi[cur][i]; }
#pragma unroll
        for (int nt = 0; nt < 4; ++nt) {
            acc[nt] = __builtin_amdgcn_wmma_f32_16x16x32_bf16(
                false, a, false, bfrag[cur][nt], (short)0, acc[nt], false, false);
        }
    }

    // epilogue: D layout VGPR j -> M = j + sel*8, N = l16
    int mOut = mTile * 16 + sel * 8;
    float bias_nt[4];
#pragma unroll
    for (int nt = 0; nt < 4; ++nt) bias_nt[nt] = bias[nBase + nt * 16 + l16];
#pragma unroll
    for (int j = 0; j < 8; ++j) {
        int m = mOut + j;
        float d = dinv[m];
        float dd = d * d;
        size_t rowoff = (size_t)m * NC + nBase + l16;
#pragma unroll
        for (int nt = 0; nt < 4; ++nt) {
            float v = acc[nt][j];
            Cl[rowoff + nt * 16]    = v;
            Cinit[rowoff + nt * 16] = dd * v + bias_nt[nt];
        }
    }
}

// ---------------- host-side orchestration ----------------

static inline int nblk(long total, int bs) { return (int)((total + bs - 1) / bs); }

extern "C" void kernel_launch(void* const* d_in, const int* in_sizes, int n_in,
                              void* d_out, int out_size, void* d_ws, size_t ws_size,
                              hipStream_t stream) {
    const int N = in_sizes[0] / IN_C;     // 50000
    const int E = in_sizes[1] / 2;        // 800000

    const float* Wc = (const float*)d_in[8];
    const float* bc = (const float*)d_in[9];
    const float* Wm = (const float*)d_in[10];
    const float* bm = (const float*)d_in[11];
    const float* Wl = (const float*)d_in[12];
    const float* bl = (const float*)d_in[13];

    // workspace carve-out (256B aligned)
    uint8_t* ws = (uint8_t*)d_ws;
    size_t off = 0;
    auto carve = [&](size_t bytes) -> uint8_t* {
        uint8_t* p = ws + off;
        off += (bytes + 255) & ~(size_t)255;
        return p;
    };
    float*  deg    = (float*)carve((size_t)N * 4);
    float*  dinv   = (float*)carve((size_t)N * 4);
    __bf16* WcB    = (__bf16*)carve((size_t)4 * IN_C * HID_C * 2);
    __bf16* WmB    = (__bf16*)carve((size_t)4 * HID_C * OUT_C * 2);
    __bf16* WlB    = (__bf16*)carve((size_t)4 * HID_C * OUT_C * 2);
    __bf16* xb     = (__bf16*)carve((size_t)N * IN_C * 2);
    float*  h_lin  = (float*)carve((size_t)N * HID_C * 4);
    float*  h      = (float*)carve((size_t)N * HID_C * 4);
    __bf16* hb     = (__bf16*)carve((size_t)N * HID_C * 2);
    float*  lin2   = (float*)carve((size_t)N * OUT_C * 4);
    (void)ws_size;

    float* out_mu = (float*)d_out;                          // [4, N, OUT_C]
    float* out_ls = (float*)d_out + (size_t)4 * N * OUT_C;  // [4, N, OUT_C]

    const int BS = 256;

    // convert all weights to bf16 once (counts are total/4 float4 groups)
    k_f32_to_bf16_v4<<<nblk((long)IN_C * HID_C, BS), BS, 0, stream>>>(
        (const float4*)Wc, (v4bf*)WcB, (long)IN_C * HID_C);
    k_f32_to_bf16_v4<<<nblk((long)HID_C * OUT_C, BS), BS, 0, stream>>>(
        (const float4*)Wm, (v4bf*)WmB, (long)HID_C * OUT_C);
    k_f32_to_bf16_v4<<<nblk((long)HID_C * OUT_C, BS), BS, 0, stream>>>(
        (const float4*)Wl, (v4bf*)WlB, (long)HID_C * OUT_C);

    const int mTiles = (N + 15) / 16;
    dim3 gemm_blk(256);
    dim3 gemm_grid1((mTiles + 7) / 8, HID_C / 64);   // 256-wide output
    dim3 gemm_grid2((mTiles + 7) / 8, OUT_C / 64);   // 128-wide output

    for (int g = 0; g < 4; ++g) {
        const float* x  = (const float*)d_in[2 * g];
        const int*   ei = (const int*)d_in[2 * g + 1];
        const int* rows = ei;          // source nodes
        const int* cols = ei + E;      // target nodes

        // --- degree normalization (shared by all 3 convs of this graph) ---
        k_zero_f32<<<nblk(N, BS), BS, 0, stream>>>(deg, N);
        k_deg_count<<<nblk(E, BS), BS, 0, stream>>>(cols, deg, E);
        k_dinv<<<nblk(N, BS), BS, 0, stream>>>(deg, dinv, N);

        // --- conv1: h = agg(x @ Wc[g]) + bc[g] ---
        long nx4 = (long)N * IN_C / 4;
        k_f32_to_bf16_v4<<<nblk(nx4, BS), BS, 0, stream>>>((const float4*)x, (v4bf*)xb, nx4);
        k_gemm_bf16_wmma<IN_C, HID_C><<<gemm_grid1, gemm_blk, 0, stream>>>(
            xb, WcB + (size_t)g * IN_C * HID_C, h_lin, h, dinv, bc + (size_t)g * HID_C, N);
        long te1 = (long)E * (HID_C / 4);
        k_agg_edges<<<nblk(te1, BS), BS, 0, stream>>>(rows, cols, dinv, h_lin, h,
                                                      te1, 6, (HID_C / 4) - 1, HID_C);

        // --- h -> bf16 for the two 256->128 convs ---
        long nh4 = (long)N * HID_C / 4;
        k_f32_to_bf16_v4<<<nblk(nh4, BS), BS, 0, stream>>>((const float4*)h, (v4bf*)hb, nh4);

        long te2 = (long)E * (OUT_C / 4);

        // --- mu = agg(h @ Wm[g]) + bm[g]  -> d_out ---
        k_gemm_bf16_wmma<HID_C, OUT_C><<<gemm_grid2, gemm_blk, 0, stream>>>(
            hb, WmB + (size_t)g * HID_C * OUT_C, lin2, out_mu + (size_t)g * N * OUT_C,
            dinv, bm + (size_t)g * OUT_C, N);
        k_agg_edges<<<nblk(te2, BS), BS, 0, stream>>>(rows, cols, dinv, lin2,
                                                      out_mu + (size_t)g * N * OUT_C,
                                                      te2, 5, (OUT_C / 4) - 1, OUT_C);

        // --- logstd = agg(h @ Wl[g]) + bl[g] -> d_out ---
        k_gemm_bf16_wmma<HID_C, OUT_C><<<gemm_grid2, gemm_blk, 0, stream>>>(
            hb, WlB + (size_t)g * HID_C * OUT_C, lin2, out_ls + (size_t)g * N * OUT_C,
            dinv, bl + (size_t)g * OUT_C, N);
        k_agg_edges<<<nblk(te2, BS), BS, 0, stream>>>(rows, cols, dinv, lin2,
                                                      out_ls + (size_t)g * N * OUT_C,
                                                      te2, 5, (OUT_C / 4) - 1, OUT_C);
    }
    (void)n_in; (void)out_size;
}
